// m2bnet_57054345560718
// MI455X (gfx1250) — matrix operand order repaired and checked by reading, not compiled, not run on hardware
//
#include <hip/hip_runtime.h>
#include <hip/hip_bf16.h>
#include <stdint.h>

// ---------------------------------------------------------------------------
// ST-GCN-18 forward for MI455X (gfx1250), wave32 + WMMA f32_16x16x32_f16.
// 32x64 block tiles (each wave: 16x32, two accumulators -> 2 WMMA per K-step).
// B tiles staged via the Tensor Data Mover (double buffered, TENSORcnt-
// synchronized, TDM pad -> 66-float row stride, conflict-free LDS reads);
// A (weight) fragments read straight from prepacked f16 global memory as
// 16-byte v8h loads in WMMA fragment order.
// ---------------------------------------------------------------------------

typedef __attribute__((ext_vector_type(16))) _Float16 v16h;
typedef __attribute__((ext_vector_type(8)))  _Float16 v8h;
typedef __attribute__((ext_vector_type(8)))  float    v8f;
typedef __attribute__((ext_vector_type(4)))  unsigned int v4u;
typedef __attribute__((ext_vector_type(8)))  int      v8i;
typedef __attribute__((ext_vector_type(4)))  int      v4i;
typedef float f4u __attribute__((ext_vector_type(4), aligned(4)));  // 4B-aligned float4

#define NB 16
#define VJ 17

// ---------------------------------------------------------------------------
// TDM: issue a 2D f32 tile load (tile_d1 rows x tile_d0 cols) into LDS.
// tensor_dim0/1 are the valid extents measured from the tile start (TDM
// zero-fills out-of-range reads -> free K padding).  pad=1 inserts 1 dword of
// LDS padding every 32 dwords (64-wide row -> 66-float stride).
// ---------------------------------------------------------------------------
__device__ __forceinline__ void tdm_load_2d_f32(unsigned lds_addr, const float* gptr,
                                                int tile_d0, int tile_d1,
                                                long long tdim0, long long tdim1,
                                                long long stride0, int pad)
{
    if (tdim1 < 0) tdim1 = 0;
    unsigned long long ga = (unsigned long long)(uintptr_t)gptr;
    v4u g0;
    g0.x = 1u;                                        // count=1, user mode, no gather
    g0.y = lds_addr;                                  // LDS byte address
    g0.z = (unsigned)(ga & 0xffffffffull);            // global_addr[31:0]
    g0.w = (unsigned)((ga >> 32) & 0x1ffffffull) | (2u << 30);  // addr[56:32] | type=2
    v8i g1;
    unsigned w0 = (2u << 16);                         // data_size = 4 bytes
    if (pad) w0 |= (1u << 20) | (4u << 22);           // pad_enable, interval=32 dwords, amount=1 dword
    g1[0] = (int)w0;
    g1[1] = (int)(((unsigned)(tdim0 & 0xffff)) << 16);                    // tensor_dim0 lo
    g1[2] = (int)(((unsigned)((tdim0 >> 16) & 0xffff)) |
                  (((unsigned)(tdim1 & 0xffff)) << 16));                  // dim0 hi | dim1 lo
    g1[3] = (int)(((unsigned)((tdim1 >> 16) & 0xffff)) |
                  (((unsigned)tile_d0) << 16));                           // dim1 hi | tile_dim0
    g1[4] = (int)((unsigned)tile_d1 & 0xffff);                            // tile_dim1 (tile_dim2=0)
    g1[5] = (int)((unsigned)(stride0 & 0xffffffffull));                   // dim0_stride lo
    g1[6] = (int)((unsigned)((stride0 >> 32) & 0xffff));                  // dim0_stride hi
    g1[7] = 0;
    v4i z4 = {0, 0, 0, 0};
    v8i z8 = {0, 0, 0, 0, 0, 0, 0, 0};
    __builtin_amdgcn_tensor_load_to_lds(g0, g1, z4, z4, z8, 0);
}

// ---------------------------------------------------------------------------
// GEMM (stride-1 columns):  Y[n][m][j] = relu?( sum_k Wp[m,k]*X[n][k][j] + bias[m] )
// Wp: prepacked f16 (M x Kp), zero-padded K.  B tile: TDM -> LDS f32 (32x64,
// 66-float row stride), double-buffered.  128 threads = 4 waves; each wave
// owns a 16x32 output slab -> 2 WMMA per K-step.
// ---------------------------------------------------------------------------
__global__ __launch_bounds__(128)
void k_gemm_tdm(const _Float16* __restrict__ Wp, const float* __restrict__ bias,
                const float* __restrict__ X, float* __restrict__ Y,
                int M, int Kp, int Kreal, int Scols, int Sin_cols, int relu)
{
    __shared__ float Bs32[2][66 * 32];

    const int tid  = threadIdx.x;
    const int lane = tid & 31;
    const int wave = tid >> 5;
    const int wM = wave >> 1, wN = wave & 1;
    const int m0 = blockIdx.y * 32;
    const int j0 = blockIdx.x * 64;
    const int lm = lane & 15, lh = lane >> 4;
    const int cb = wN * 32;                            // wave's column base in tile

    const float* Xn = X + (size_t)blockIdx.z * Kreal * Sin_cols;
    float*       Yn = Y + (size_t)blockIdx.z * M * Scols;
    const _Float16* Arow = Wp + (size_t)(m0 + wM * 16 + lm) * Kp;

    unsigned lb0 = (unsigned)(uintptr_t)&Bs32[0][0];
    unsigned lb1 = (unsigned)(uintptr_t)&Bs32[1][0];

    const int niter = Kp >> 5;
    if (wave == 0)
        tdm_load_2d_f32(lb0, Xn + j0, 64, 32,
                        (long long)Sin_cols - j0, (long long)Kreal,
                        (long long)Sin_cols, 1);

    v8f acc0 = {}, acc1 = {};
    for (int it = 0; it < niter; ++it) {
        const int k0 = it << 5;
        if (wave == 0) {
            if (it + 1 < niter) {
                tdm_load_2d_f32((it & 1) ? lb0 : lb1,
                                Xn + (size_t)(k0 + 32) * Sin_cols + j0, 64, 32,
                                (long long)Sin_cols - j0, (long long)(Kreal - (k0 + 32)),
                                (long long)Sin_cols, 1);
                __builtin_amdgcn_s_wait_tensorcnt((short)1);   // tile `it` has landed
            } else {
                __builtin_amdgcn_s_wait_tensorcnt((short)0);
            }
        }
        __syncthreads();

        // A fragment: two 16B global loads from prepacked f16 weights
        v8h a0 = *(const v8h*)(Arow + k0 + lh * 16);
        v8h a1 = *(const v8h*)(Arow + k0 + lh * 16 + 8);
        v16h a = __builtin_shufflevector(a0, a1, 0, 1, 2, 3, 4, 5, 6, 7,
                                         8, 9, 10, 11, 12, 13, 14, 15);
        // Two B fragments from padded f32 LDS tile (row stride 66, pad = wN)
        const float* Bc = &Bs32[it & 1][0];
        const int c0 = cb + lm + wN;        // pad offset (col>>5) == wN for both halves
        const int c1 = cb + 16 + lm + wN;
        v16h b0, b1;
#pragma unroll
        for (int e = 0; e < 16; ++e) {
            int row = (lh * 16 + e) * 66;
            b0[e] = (_Float16)Bc[row + c0];
            b1[e] = (_Float16)Bc[row + c1];
        }

        acc0 = __builtin_amdgcn_wmma_f32_16x16x32_f16(false, a, false, b0,
                                                      (short)0, acc0, false, false);
        acc1 = __builtin_amdgcn_wmma_f32_16x16x32_f16(false, a, false, b1,
                                                      (short)0, acc1, false, false);
        __syncthreads();
    }

    const int col0 = j0 + cb + lm;
#pragma unroll
    for (int r = 0; r < 8; ++r) {
        int m = m0 + wM * 16 + lh * 8 + r;
        float v0 = acc0[r] + bias[m];
        float v1 = acc1[r] + bias[m];
        if (relu) { v0 = fmaxf(v0, 0.0f); v1 = fmaxf(v1, 0.0f); }
        Yn[(size_t)m * Scols + col0]      = v0;
        Yn[(size_t)m * Scols + col0 + 16] = v1;
    }
}

// ---------------------------------------------------------------------------
// Temporal conv (9 taps, stride st) as WMMA GEMM, K-loop over (dt, cin).
// Wp: prepacked f16 [9][C][C] (dt-major -> contiguous A fragments from global).
// B tile (32x64) staged manually (temporal shift + zero pad), f16 transposed
// LDS layout [col][k] with 48-half row stride -> fragments are ds_load_b128.
// Each wave: 16x32 slab -> 2 WMMA per (dt, k-chunk).
// ---------------------------------------------------------------------------
__global__ __launch_bounds__(128)
void k_tcn_wmma(const _Float16* __restrict__ Wp, const float* __restrict__ bias,
                const float* __restrict__ X, float* __restrict__ Y,
                int C, int Ti, int To, int st)
{
    __shared__ _Float16 Bt[64 * 48];

    const int tid  = threadIdx.x;
    const int lane = tid & 31;
    const int wave = tid >> 5;
    const int wM = wave >> 1, wN = wave & 1;
    const int m0 = blockIdx.y * 32;
    const int j0 = blockIdx.x * 64;
    const int lm = lane & 15, lh = lane >> 4;
    const int cb = wN * 32;

    const float* Xn = X + (size_t)blockIdx.z * C * Ti * VJ;
    float*       Yn = Y + (size_t)blockIdx.z * C * To * VJ;
    const int arow = m0 + wM * 16 + lm;

    v8f acc0 = {}, acc1 = {};
    for (int dt = 0; dt < 9; ++dt) {
        const _Float16* Wd = Wp + ((size_t)dt * C + arow) * C;
        for (int k0 = 0; k0 < C; k0 += 32) {
            // ---- stage B transposed: Bt[col][k], vector loads for interior quads ----
#pragma unroll
            for (int q8 = 0; q8 < 4; ++q8) {
                int q  = q8 * 128 + tid;          // 0..511 quad id
                int r  = q >> 4;                  // k row 0..31
                int c4 = (q & 15) * 4;            // tile col start 0..60
                int gj = j0 + c4;
                int t0 = gj / VJ, v0 = gj - t0 * VJ;
                int t3 = (gj + 3) / VJ;
                int ti0 = t0 * st + dt - 4;
                f4u vv;
                if (t0 == t3 && ti0 >= 0 && ti0 < Ti) {
                    vv = *(const f4u*)(Xn + ((size_t)(k0 + r) * Ti + ti0) * VJ + v0);
                } else {
#pragma unroll
                    for (int u = 0; u < 4; ++u) {
                        int gjj = gj + u;
                        int tt = gjj / VJ, vq = gjj - tt * VJ;
                        int tii = tt * st + dt - 4;
                        vv[u] = (tii >= 0 && tii < Ti)
                                  ? Xn[((size_t)(k0 + r) * Ti + tii) * VJ + vq] : 0.0f;
                    }
                }
                Bt[(c4 + 0) * 48 + r] = (_Float16)vv.x;
                Bt[(c4 + 1) * 48 + r] = (_Float16)vv.y;
                Bt[(c4 + 2) * 48 + r] = (_Float16)vv.z;
                Bt[(c4 + 3) * 48 + r] = (_Float16)vv.w;
            }
            __syncthreads();

            v8h a0 = *(const v8h*)(Wd + k0 + lh * 16);
            v8h a1 = *(const v8h*)(Wd + k0 + lh * 16 + 8);
            v16h a = __builtin_shufflevector(a0, a1, 0, 1, 2, 3, 4, 5, 6, 7,
                                             8, 9, 10, 11, 12, 13, 14, 15);
            const _Float16* bc0 = &Bt[(cb + lm) * 48 + lh * 16];
            const _Float16* bc1 = &Bt[(cb + 16 + lm) * 48 + lh * 16];
            v16h b0 = __builtin_shufflevector(*(const v8h*)bc0, *(const v8h*)(bc0 + 8),
                                              0, 1, 2, 3, 4, 5, 6, 7,
                                              8, 9, 10, 11, 12, 13, 14, 15);
            v16h b1 = __builtin_shufflevector(*(const v8h*)bc1, *(const v8h*)(bc1 + 8),
                                              0, 1, 2, 3, 4, 5, 6, 7,
                                              8, 9, 10, 11, 12, 13, 14, 15);

            acc0 = __builtin_amdgcn_wmma_f32_16x16x32_f16(false, a, false, b0,
                                                          (short)0, acc0, false, false);
            acc1 = __builtin_amdgcn_wmma_f32_16x16x32_f16(false, a, false, b1,
                                                          (short)0, acc1, false, false);
            __syncthreads();
        }
    }

    const int col0 = j0 + cb + lm;
#pragma unroll
    for (int r = 0; r < 8; ++r) {
        int m = m0 + wM * 16 + lh * 8 + r;
        float bm = bias[m];
        Yn[(size_t)m * (To * VJ) + col0]      = acc0[r] + bm;
        Yn[(size_t)m * (To * VJ) + col0 + 16] = acc1[r] + bm;
    }
}

// ---------------------------------------------------------------------------
// Weight repack kernels (fp32 -> f16, zero-padded / dt-major for tcn)
// ---------------------------------------------------------------------------
__global__ __launch_bounds__(256)
void k_pack_w(const float* __restrict__ W, _Float16* __restrict__ P,
              int M, int K, int Kp)
{
    int i = blockIdx.x * 256 + threadIdx.x;
    if (i >= M * Kp) return;
    int m = i / Kp, k = i - m * Kp;
    P[i] = (k < K) ? (_Float16)W[(size_t)m * K + k] : (_Float16)0.0f;
}

__global__ __launch_bounds__(256)
void k_pack_tcn(const float* __restrict__ W, _Float16* __restrict__ P, int C)
{
    int i = blockIdx.x * 256 + threadIdx.x;
    if (i >= 9 * C * C) return;
    int dt = i / (C * C), r = i - dt * (C * C);
    int m = r / C, k = r - m * C;
    P[i] = (_Float16)W[((size_t)m * C + k) * 9 + dt];
}

// ---------------------------------------------------------------------------
// Residual 1x1 conv with temporal stride (tiny: <2% of FLOPs) -- plain VALU.
// ---------------------------------------------------------------------------
__global__ __launch_bounds__(256)
void k_res_conv(const float* __restrict__ W, const float* __restrict__ bias,
                const float* __restrict__ X, float* __restrict__ Y,
                int C, int Cin, int Ti, int To, int st)
{
    long long total = (long long)NB * C * To * VJ;
    long long idx = (long long)blockIdx.x * 256 + threadIdx.x;
    if (idx >= total) return;
    int j = (int)(idx % (To * VJ));
    long long r = idx / (To * VJ);
    int m = (int)(r % C);
    int n = (int)(r / C);
    int col = (j / VJ) * st * VJ + (j % VJ);
    const float* Xn = X + (size_t)n * Cin * Ti * VJ;
    float s = bias[m];
    for (int k = 0; k < Cin; ++k)
        s += W[(size_t)m * Cin + k] * Xn[(size_t)k * Ti * VJ + col];
    Y[idx] = s;
}

// ---------------------------------------------------------------------------
// data_bn: x (N,T,17,3); per-(v,c) stats over (N,T); emit h (N,3,T,17).
// ---------------------------------------------------------------------------
__global__ __launch_bounds__(256)
void k_data_bn(const float* __restrict__ X, const float* __restrict__ g,
               const float* __restrict__ b, float* __restrict__ H)
{
    const int ch = blockIdx.x;          // = v*3 + c
    const int v = ch / 3, c = ch % 3;
    __shared__ float s1[256], s2[256];
    float a = 0.f, a2 = 0.f;
    for (int i = threadIdx.x; i < NB * 1024; i += 256) {
        float val = X[(size_t)i * 51 + ch];
        a += val; a2 += val * val;
    }
    s1[threadIdx.x] = a; s2[threadIdx.x] = a2;
    __syncthreads();
    for (int off = 128; off > 0; off >>= 1) {
        if (threadIdx.x < off) {
            s1[threadIdx.x] += s1[threadIdx.x + off];
            s2[threadIdx.x] += s2[threadIdx.x + off];
        }
        __syncthreads();
    }
    const float cnt  = (float)(NB * 1024);
    const float mean = s1[0] / cnt;
    const float var  = s2[0] / cnt - mean * mean;
    const float rs   = rsqrtf(var + 1e-5f);
    const float gg   = g[ch] * rs;
    const float bb   = b[ch] - mean * gg;
    for (int i = threadIdx.x; i < NB * 1024; i += 256) {
        int n = i >> 10, t = i & 1023;
        float val = X[(size_t)i * 51 + ch];
        H[(((size_t)n * 3 + c) * 1024 + t) * VJ + v] = val * gg + bb;
    }
}

// ---------------------------------------------------------------------------
// Per-channel BN stats over (N, S) for (N, C, S). mr[c]=mean, mr[C+c]=rstd.
// ---------------------------------------------------------------------------
__global__ __launch_bounds__(256)
void k_bn_stats(const float* __restrict__ X, float* __restrict__ mr,
                int C, int S, int Nbatch)
{
    const int c = blockIdx.x;
    __shared__ float s1[256], s2[256];
    float a = 0.f, a2 = 0.f;
    for (int n = 0; n < Nbatch; ++n) {
        const float* p = X + ((size_t)n * C + c) * S;
        for (int i = threadIdx.x; i < S; i += 256) {
            float v = p[i]; a += v; a2 += v * v;
        }
    }
    s1[threadIdx.x] = a; s2[threadIdx.x] = a2;
    __syncthreads();
    for (int off = 128; off > 0; off >>= 1) {
        if (threadIdx.x < off) {
            s1[threadIdx.x] += s1[threadIdx.x + off];
            s2[threadIdx.x] += s2[threadIdx.x + off];
        }
        __syncthreads();
    }
    if (threadIdx.x == 0) {
        float cnt  = (float)Nbatch * (float)S;
        float mean = s1[0] / cnt;
        float var  = s2[0] / cnt - mean * mean;
        mr[c]     = mean;
        mr[C + c] = rsqrtf(var + 1e-5f);
    }
}

// ---------------------------------------------------------------------------
// BN apply (+ optional residual, + optional ReLU). X layout (N,C,S).
// ---------------------------------------------------------------------------
__global__ __launch_bounds__(256)
void k_bn_apply(const float* __restrict__ X, const float* __restrict__ mr,
                const float* __restrict__ g, const float* __restrict__ b,
                const float* __restrict__ res, float* __restrict__ Y,
                int C, int S, long long total, int relu)
{
    long long idx = (long long)blockIdx.x * 256 + threadIdx.x;
    if (idx >= total) return;
    int c = (int)((idx / S) % C);
    float v = (X[idx] - mr[c]) * mr[C + c] * g[c] + b[c];
    if (res) v += res[idx];
    if (relu) v = fmaxf(v, 0.0f);
    Y[idx] = v;
}

// ---------------------------------------------------------------------------
// Graph einsum: Z[n,c,t,w] = sum_k sum_v Y[n,k*C+c,t,v] * (A*ei)[k,v,w]
// ---------------------------------------------------------------------------
__global__ __launch_bounds__(256)
void k_gcn_einsum(const float* __restrict__ Y, const float* __restrict__ A,
                  const float* __restrict__ EI, float* __restrict__ Z,
                  int C, int Tlen, int Nbatch)
{
    __shared__ float Aei[3 * VJ * VJ];
    for (int i = threadIdx.x; i < 3 * VJ * VJ; i += 256) Aei[i] = A[i] * EI[i];
    __syncthreads();
    long long total = (long long)Nbatch * C * Tlen * VJ;
    long long idx = (long long)blockIdx.x * 256 + threadIdx.x;
    if (idx >= total) return;
    int w = (int)(idx % VJ);
    long long r = idx / VJ;
    int t = (int)(r % Tlen); r /= Tlen;
    int c = (int)(r % C);
    int n = (int)(r / C);
    float s = 0.f;
#pragma unroll
    for (int k = 0; k < 3; ++k) {
        const float* yb = Y + (((size_t)(n * 3 * C + k * C + c)) * Tlen + t) * VJ;
        const float* ab = Aei + k * (VJ * VJ) + w;
#pragma unroll
        for (int v = 0; v < VJ; ++v) s += yb[v] * ab[v * VJ];
    }
    Z[idx] = s;
}

// ---------------------------------------------------------------------------
__global__ __launch_bounds__(256)
void k_pool_v(const float* __restrict__ X, float* __restrict__ Y, long long total)
{
    long long idx = (long long)blockIdx.x * 256 + threadIdx.x;
    if (idx >= total) return;
    const float* p = X + idx * VJ;
    float s = 0.f;
#pragma unroll
    for (int v = 0; v < VJ; ++v) s += p[v];
    Y[idx] = s * (1.0f / (float)VJ);
}

__global__ __launch_bounds__(256)
void k_gather_seq(const float* __restrict__ F, float* __restrict__ L, int Ntok)
{
    int idx = blockIdx.x * 256 + threadIdx.x;
    if (idx >= 128 * Ntok) return;
    int tok = idx % Ntok;
    int d   = idx / Ntok;
    int n = tok >> 10, tt = tok & 1023;
    int tc = tt >> 2, q = tt & 3;
    L[idx] = F[((size_t)(n * 512 + q * 128 + d)) * 256 + tc];
}

__global__ __launch_bounds__(256)
void k_proj(const float* __restrict__ X, const float* __restrict__ w,
            const float* __restrict__ b, float* __restrict__ out, int Ntok)
{
    int tok = blockIdx.x * 256 + threadIdx.x;
    if (tok >= Ntok) return;
    float s = b[0];
#pragma unroll 4
    for (int d = 0; d < 128; ++d) s += X[(size_t)d * Ntok + tok] * w[d];
    out[tok] = s;
}

// ---------------------------------------------------------------------------
// Host orchestration
// ---------------------------------------------------------------------------
struct Cfg { int cin, cout, stride, residual; };
static const Cfg g_cfg[10] = {
    {3, 64, 1, 0}, {64, 64, 1, 1}, {64, 64, 1, 1}, {64, 64, 1, 1},
    {64, 128, 2, 1}, {128, 128, 1, 1}, {128, 128, 1, 1},
    {128, 256, 2, 1}, {256, 256, 1, 1}, {256, 256, 1, 1}};

struct BP {
    const float *bn1_b, *bn1_g, *bn2_b, *bn2_g, *ei, *gcn_b, *gcn_w;
    const float *res_b, *res_be, *res_g, *res_w, *tcn_b, *tcn_w;
};

extern "C" void kernel_launch(void* const* d_in, const int* in_sizes, int n_in,
                              void* d_out, int out_size, void* d_ws, size_t ws_size,
                              hipStream_t stream)
{
    (void)in_sizes; (void)n_in; (void)out_size; (void)ws_size;

    // ---- input pointer mapping (JAX pytree leaf order: sorted dict keys) ----
    const float* x    = (const float*)d_in[0];
    const float* Amat = (const float*)d_in[1];
    int idx = 2;
    BP bp[10];
    for (int b = 0; b < 10; ++b) {
        bp[b].bn1_b = (const float*)d_in[idx++];
        bp[b].bn1_g = (const float*)d_in[idx++];
        bp[b].bn2_b = (const float*)d_in[idx++];
        bp[b].bn2_g = (const float*)d_in[idx++];
        bp[b].ei    = (const float*)d_in[idx++];
        bp[b].gcn_b = (const float*)d_in[idx++];
        bp[b].gcn_w = (const float*)d_in[idx++];
        bool hasres = g_cfg[b].residual &&
                      (g_cfg[b].cin != g_cfg[b].cout || g_cfg[b].stride != 1);
        if (hasres) {
            bp[b].res_b  = (const float*)d_in[idx++];
            bp[b].res_be = (const float*)d_in[idx++];
            bp[b].res_g  = (const float*)d_in[idx++];
            bp[b].res_w  = (const float*)d_in[idx++];
        } else {
            bp[b].res_b = bp[b].res_be = bp[b].res_g = bp[b].res_w = nullptr;
        }
        bp[b].tcn_b = (const float*)d_in[idx++];
        bp[b].tcn_w = (const float*)d_in[idx++];
    }
    const float* data_bn_b = (const float*)d_in[idx++];
    const float* data_bn_g = (const float*)d_in[idx++];
    const float* fcn_b     = (const float*)d_in[idx++];
    const float* fcn_w     = (const float*)d_in[idx++];
    const float* lin1_b    = (const float*)d_in[idx++];
    const float* lin1_w    = (const float*)d_in[idx++];
    const float* lin2_b    = (const float*)d_in[idx++];
    const float* lin2_w    = (const float*)d_in[idx++];
    const float* proj_b    = (const float*)d_in[idx++];
    const float* proj_w    = (const float*)d_in[idx++];

    // ---- workspace bump allocation ----
    char* base = (char*)d_ws;
    size_t off = 0;
    auto alloc = [&](size_t bytes) -> void* {
        void* p = (void*)(base + off);
        off = (off + bytes + 255) & ~(size_t)255;
        return p;
    };
    const size_t ACT = (size_t)NB * 64 * 1024 * VJ;   // 17,825,792 f32 per stage
    float* act0 = (float*)alloc(ACT * 4);
    float* act1 = (float*)alloc(ACT * 4);
    float* bufZ = (float*)alloc(ACT * 4);
    float* bufT = (float*)alloc(ACT * 4);
    float* bufR = (float*)alloc(ACT * 4);
    float* bufY = (float*)alloc(ACT * 3 * 4);
    float* mr1  = (float*)alloc(512 * 4);
    float* mr2  = (float*)alloc(512 * 4);
    float* mrR  = (float*)alloc(512 * 4);
    float* pooled = (float*)alloc((size_t)NB * 256 * 256 * 4);
    float* fcnout = (float*)alloc((size_t)NB * 512 * 256 * 4);
    float* seqin  = (float*)alloc((size_t)128 * 16384 * 4);
    float* lin1o  = (float*)alloc((size_t)256 * 16384 * 4);
    float* lin2o  = (float*)alloc((size_t)128 * 16384 * 4);

    // packed f16 weights
    _Float16* pw_gcn[10];
    _Float16* pw_tcn[10];
    int       kp_gcn[10];
    for (int b = 0; b < 10; ++b) {
        int C = g_cfg[b].cout, Cin = g_cfg[b].cin;
        kp_gcn[b] = (Cin + 31) & ~31;
        pw_gcn[b] = (_Float16*)alloc((size_t)3 * C * kp_gcn[b] * 2);
        pw_tcn[b] = (_Float16*)alloc((size_t)9 * C * C * 2);
    }
    _Float16* pw_fcn = (_Float16*)alloc((size_t)512 * 256 * 2);
    _Float16* pw_l1  = (_Float16*)alloc((size_t)256 * 128 * 2);
    _Float16* pw_l2  = (_Float16*)alloc((size_t)128 * 256 * 2);

    // ---- repack weights to f16 ----
    for (int b = 0; b < 10; ++b) {
        int C = g_cfg[b].cout, Cin = g_cfg[b].cin;
        int ng = 3 * C * kp_gcn[b];
        k_pack_w<<<(ng + 255) / 256, 256, 0, stream>>>(bp[b].gcn_w, pw_gcn[b],
                                                       3 * C, Cin, kp_gcn[b]);
        int nt = 9 * C * C;
        k_pack_tcn<<<(nt + 255) / 256, 256, 0, stream>>>(bp[b].tcn_w, pw_tcn[b], C);
    }
    k_pack_w<<<(512 * 256 + 255) / 256, 256, 0, stream>>>(fcn_w, pw_fcn, 512, 256, 256);
    k_pack_w<<<(256 * 128 + 255) / 256, 256, 0, stream>>>(lin1_w, pw_l1, 256, 128, 128);
    k_pack_w<<<(128 * 256 + 255) / 256, 256, 0, stream>>>(lin2_w, pw_l2, 128, 256, 256);

    // ---- data BN: x (N,T,17,3) -> act0 (N,3,T,17) ----
    k_data_bn<<<51, 256, 0, stream>>>(x, data_bn_g, data_bn_b, act0);

    float* cur = act0;
    float* nxt = act1;
    int Tcur = 1024;

    for (int b = 0; b < 10; ++b) {
        const int Cin = g_cfg[b].cin, C = g_cfg[b].cout, st = g_cfg[b].stride;
        const int Ti = Tcur, To = Ti / st;
        const int Sin = Ti * VJ, Sout = To * VJ;

        // spatial 1x1 conv (WMMA + TDM) -> (N, 3C, Ti, 17)
        dim3 g1(Sin / 64, (3 * C) / 32, NB);
        k_gemm_tdm<<<g1, 128, 0, stream>>>(pw_gcn[b], bp[b].gcn_b, cur, bufY,
                                           3 * C, kp_gcn[b], Cin, Sin, Sin, 0);

        // einsum with A*ei -> bufZ
        long long tot1 = (long long)NB * C * Ti * VJ;
        k_gcn_einsum<<<(unsigned)((tot1 + 255) / 256), 256, 0, stream>>>(
            bufY, Amat, bp[b].ei, bufZ, C, Ti, NB);

        // bn1 + ReLU (in place)
        k_bn_stats<<<C, 256, 0, stream>>>(bufZ, mr1, C, Sin, NB);
        k_bn_apply<<<(unsigned)((tot1 + 255) / 256), 256, 0, stream>>>(
            bufZ, mr1, bp[b].bn1_g, bp[b].bn1_b, nullptr, bufZ, C, Sin, tot1, 1);

        // temporal conv (WMMA) -> bufT
        dim3 g2(Sout / 64, C / 32, NB);
        k_tcn_wmma<<<g2, 128, 0, stream>>>(pw_tcn[b], bp[b].tcn_b, bufZ, bufT,
                                           C, Ti, To, st);

        long long tot2 = (long long)NB * C * To * VJ;
        k_bn_stats<<<C, 256, 0, stream>>>(bufT, mr2, C, Sout, NB);

        // residual path
        const float* res = nullptr;
        if (g_cfg[b].residual) {
            if (Cin == C && st == 1) {
                res = cur;
            } else {
                k_res_conv<<<(unsigned)((tot2 + 255) / 256), 256, 0, stream>>>(
                    bp[b].res_w, bp[b].res_b, cur, bufR, C, Cin, Ti, To, st);
                k_bn_stats<<<C, 256, 0, stream>>>(bufR, mrR, C, Sout, NB);
                k_bn_apply<<<(unsigned)((tot2 + 255) / 256), 256, 0, stream>>>(
                    bufR, mrR, bp[b].res_g, bp[b].res_be, nullptr, bufR, C, Sout, tot2, 0);
                res = bufR;
            }
        }

        // bn2 + residual + ReLU -> next activation
        k_bn_apply<<<(unsigned)((tot2 + 255) / 256), 256, 0, stream>>>(
            bufT, mr2, bp[b].bn2_g, bp[b].bn2_b, res, nxt, C, Sout, tot2, 1);

        float* tmp = cur; cur = nxt; nxt = tmp;
        Tcur = To;
    }

    // ---- head: pool joints, fcn, reshape, MLP, projection ----
    long long pt = (long long)NB * 256 * 256;
    k_pool_v<<<(unsigned)((pt + 255) / 256), 256, 0, stream>>>(cur, pooled, pt);

    dim3 gf(256 / 64, 512 / 32, NB);
    k_gemm_tdm<<<gf, 128, 0, stream>>>(pw_fcn, fcn_b, pooled, fcnout,
                                       512, 256, 256, 256, 256, 0);

    k_gather_seq<<<(128 * 16384) / 256, 256, 0, stream>>>(fcnout, seqin, 16384);

    dim3 gl1(16384 / 64, 256 / 32, 1);
    k_gemm_tdm<<<gl1, 128, 0, stream>>>(pw_l1, lin1_b, seqin, lin1o,
                                        256, 128, 128, 16384, 16384, 1);
    dim3 gl2(16384 / 64, 128 / 32, 1);
    k_gemm_tdm<<<gl2, 128, 0, stream>>>(pw_l2, lin2_b, lin1o, lin2o,
                                        128, 256, 256, 16384, 16384, 1);

    k_proj<<<16384 / 256, 256, 0, stream>>>(lin2o, proj_w, proj_b,
                                            (float*)d_out, 16384);
}
